// H2GFormerLayer_90443421319562
// MI455X (gfx1250) — matrix operand order repaired and checked by reading, not compiled, not run on hardware
//
#include <hip/hip_runtime.h>
#include <hip/hip_bf16.h>
#include <math.h>

// ---------------- problem constants (must match reference) ----------------
#define T_NT   2
#define ET_N   2
#define NPER   25000
#define EPER   200000
#define HID    256
#define HID2   512
#define HEADS  8
#define NTOT   (T_NT * NPER)
#define LN_EPS 1e-5f

typedef __attribute__((ext_vector_type(16))) _Float16 v16h;
typedef __attribute__((ext_vector_type(8)))  _Float16 h8;
typedef __attribute__((ext_vector_type(8)))  float    v8f;

// ------------------------ vector load/store helpers ------------------------
__device__ __forceinline__ void ld8(const float* __restrict__ p, float* u) {
  float4 a = *(const float4*)p;
  float4 b = *(const float4*)(p + 4);
  u[0]=a.x; u[1]=a.y; u[2]=a.z; u[3]=a.w; u[4]=b.x; u[5]=b.y; u[6]=b.z; u[7]=b.w;
}
__device__ __forceinline__ void st8(float* __restrict__ p, const float* u) {
  float4 a = {u[0],u[1],u[2],u[3]};
  float4 b = {u[4],u[5],u[6],u[7]};
  *(float4*)p = a; *(float4*)(p + 4) = b;
}

// ---------------------------- tiny utility kernels ------------------------
__global__ __launch_bounds__(256) void zero_kernel(float* __restrict__ p, long n) {
  long i = (long)blockIdx.x * 256 + threadIdx.x;
  if (i < n) p[i] = 0.0f;
}

// degree counts for both etypes in one pass
__global__ __launch_bounds__(256) void deg_kernel(const int* __restrict__ esrc,
                                                  const int* __restrict__ edst,
                                                  float* __restrict__ cs,
                                                  float* __restrict__ cd) {
  long i = (long)blockIdx.x * 256 + threadIdx.x;
  if (i >= (long)ET_N * EPER) return;
  int et = (int)(i / EPER);
  atomicAdd(&cs[(size_t)et * NPER + esrc[i]], 1.0f);
  atomicAdd(&cd[(size_t)et * NPER + edst[i]], 1.0f);
}

__global__ __launch_bounds__(256) void invsqrt_kernel(float* __restrict__ p, long n) {
  long i = (long)blockIdx.x * 256 + threadIdx.x;
  if (i < n) p[i] = rsqrtf(fmaxf(p[i], 1.0f));
}

// weight pre-pass: W[B,K,N] f32 row-major -> WT[B,N,K] f16 (K-contiguous rows)
__global__ __launch_bounds__(256) void wtrans_kernel(const float* __restrict__ W,
                                                     _Float16* __restrict__ WT,
                                                     int K, int N, int B) {
  long i = (long)blockIdx.x * 256 + threadIdx.x;
  long KN = (long)K * N;
  if (i >= (long)B * KN) return;
  int t = (int)(i / KN);
  long r = i - (long)t * KN;
  int k = (int)(r / N), n = (int)(r - (long)k * N);
  WT[(size_t)t * KN + (size_t)n * K + k] = (_Float16)W[i];
}

// ------------------------------- LayerNorm --------------------------------
// one wave per 256-wide row, lane owns contiguous 8 floats (float4 x2)
__global__ __launch_bounds__(256) void ln_kernel(const float* __restrict__ x,
                                                 const float* __restrict__ g,
                                                 const float* __restrict__ b,
                                                 float* __restrict__ y) {
  int row  = blockIdx.x * 8 + (threadIdx.x >> 5);
  if (row >= NTOT) return;
  int lane = threadIdx.x & 31;
  const float* xp = x + (size_t)row * HID + lane * 8;
  float v[8];
  ld8(xp, v);
  float s = 0.0f;
#pragma unroll
  for (int j = 0; j < 8; ++j) s += v[j];
#pragma unroll
  for (int m = 16; m > 0; m >>= 1) s += __shfl_xor(s, m, 32);
  float mu = s * (1.0f / HID);
  float vs = 0.0f;
#pragma unroll
  for (int j = 0; j < 8; ++j) { float d = v[j] - mu; vs += d * d; }
#pragma unroll
  for (int m = 16; m > 0; m >>= 1) vs += __shfl_xor(vs, m, 32);
  float rstd = rsqrtf(vs * (1.0f / HID) + LN_EPS);
  int t = (row >= NPER) ? 1 : 0;
  float gg[8], bb[8];
  ld8(g + (size_t)t * HID + lane * 8, gg);
  ld8(b + (size_t)t * HID + lane * 8, bb);
  float o[8];
#pragma unroll
  for (int j = 0; j < 8; ++j) o[j] = (v[j] - mu) * rstd * gg[j] + bb[j];
  st8(y + (size_t)row * HID + lane * 8, o);
}

// --------------------- graph-conv scatter (segment_sum) -------------------
__global__ __launch_bounds__(256) void conv_scatter_kernel(const int* __restrict__ esrc,
                                                           const int* __restrict__ edst,
                                                           const float* __restrict__ ln,
                                                           const float* __restrict__ cs,
                                                           float* __restrict__ agg) {
  long e = (long)blockIdx.x * 8 + (threadIdx.x >> 5);
  if (e >= (long)ET_N * EPER) return;
  int lane = threadIdx.x & 31;
  int et = (int)(e / EPER);
  int s = esrc[e], d = edst[e];
  int st = et, dt = 1 - et;      // ETYPE_MAP = ((0,1),(1,0))
  float c = cs[(size_t)et * NPER + s];
  float v[8];
  ld8(ln + ((size_t)st * NPER + s) * HID + lane * 8, v);
  float* dp = agg + ((size_t)dt * NPER + d) * HID + lane * 8;
#pragma unroll
  for (int j = 0; j < 8; ++j) atomicAdd(&dp[j], v[j] * c);
}

// ----------------------- attention: scores -> exp/denom --------------------
// Fixed-shift softmax (M=5 >= clipped score max) is exactly the reference
// softmax: per-dst constant shift cancels in alpha.
// lane owns 8 contiguous dims -> exactly one head (head = lane>>2).
__global__ __launch_bounds__(256) void attn_score_kernel(const int* __restrict__ esrc,
                                                         const int* __restrict__ edst,
                                                         const float* __restrict__ q,
                                                         const float* __restrict__ k,
                                                         float* __restrict__ ex,
                                                         float* __restrict__ denom,
                                                         int src_off, int dst_off) {
  int e = blockIdx.x * 8 + (threadIdx.x >> 5);
  if (e >= EPER) return;
  int lane = threadIdx.x & 31;
  int sg = esrc[e] + src_off;
  int dg = edst[e] + dst_off;
  float qv[8], kv[8];
  ld8(q + (size_t)dg * HID + lane * 8, qv);
  ld8(k + (size_t)sg * HID + lane * 8, kv);
  float s = 0.0f;
#pragma unroll
  for (int j = 0; j < 8; ++j) s += qv[j] * kv[j];
  s += __shfl_xor(s, 1, 32);
  s += __shfl_xor(s, 2, 32);     // sum over the 4 lanes of this head
  if ((lane & 3) == 0) {
    int head = lane >> 2;
    float sc = s * 0.17677669529663687f;          // 1/sqrt(32)
    sc = fminf(5.0f, fmaxf(-5.0f, sc));
    float ev = expf(sc - 5.0f);
    ex[(size_t)e * HEADS + head] = ev;
    atomicAdd(&denom[(size_t)dg * HEADS + head], ev);
  }
}

// out[dst] += alpha * v[src]; lane's 8 dims share one head -> one alpha
__global__ __launch_bounds__(256) void attn_agg_kernel(const int* __restrict__ esrc,
                                                       const int* __restrict__ edst,
                                                       const float* __restrict__ ex,
                                                       const float* __restrict__ denom,
                                                       const float* __restrict__ v,
                                                       float* __restrict__ outp,
                                                       int src_off, int dst_off) {
  int e = blockIdx.x * 8 + (threadIdx.x >> 5);
  if (e >= EPER) return;
  int lane = threadIdx.x & 31;
  int sg = esrc[e] + src_off;
  int dg = edst[e] + dst_off;
  int head = lane >> 2;
  float al = ex[(size_t)e * HEADS + head] / denom[(size_t)dg * HEADS + head];
  float vv[8];
  ld8(v + (size_t)sg * HID + lane * 8, vv);
  float* op = outp + (size_t)dg * HID + lane * 8;
#pragma unroll
  for (int j = 0; j < 8; ++j) atomicAdd(&op[j], al * vv[j]);
}

// ------------------------------ WMMA GEMM ---------------------------------
// C[M,N] = A[M,K] @ W[K,N] (+epilogue). A: f32, converted in-register.
// W: pre-transposed f16 WT[N,K] so each lane's B fragment is K-contiguous.
// One wave per 16x64 tile (A fragment reused across 4 WMMAs per K-chunk).
// Fragment layouts per CDNA5 ISA 7.12.2 (wave32):
//  A 16x32: lanes 0-15 row=lane&15, K={0..7,16..23}; lanes 16-31 K+8
//  B 32x16: col=lane&15; lanes 0-15 K={0..7,16..23}; lanes 16-31 K+8
//  C 16x16: VGPR r -> M = r + 8*(lane>=16), N = lane&15
__device__ __forceinline__ v16h load_bfrag(const _Float16* __restrict__ w) {
  h8 x0 = *(const h8*)w;
  h8 x1 = *(const h8*)(w + 16);
  v16h r;
#pragma unroll
  for (int i = 0; i < 8; ++i) { r[i] = x0[i]; r[i + 8] = x1[i]; }
  return r;
}

// MODE: 0=acc+bias ; 1=base+acc+bias ; 2=base+acc*rowscale+bias ; 3=gelu(acc+bias)
template <int MODE>
__global__ __launch_bounds__(128) void gemm_kernel(const float* __restrict__ A,
                                                   const _Float16* __restrict__ WT,
                                                   const float* __restrict__ bias,
                                                   const float* __restrict__ base,
                                                   const float* __restrict__ rowscale,
                                                   float* __restrict__ out,
                                                   int M, int K, int N,
                                                   int Ngrp, int total_tiles) {
  int wid = blockIdx.x * 4 + (threadIdx.x >> 5);
  if (wid >= total_tiles) return;
  int lane = threadIdx.x & 31;
  int mt = wid / Ngrp, ng = wid - mt * Ngrp;
  int row0 = mt * 16, col0 = ng * 64;

  int arow = row0 + (lane & 15);
  if (arow >= M) arow = M - 1;                 // clamp (stores guarded)
  const float* ap = A + (size_t)arow * K;
  int khalf = (lane >> 4) * 8;                 // A K-shift for hi half-lanes
  int colL  = lane & 15;
  int kbB   = (lane >> 4) * 8;                 // B K-shift for hi half-lanes

  const _Float16* wt0 = WT + (size_t)(col0 +  0 + colL) * K;
  const _Float16* wt1 = WT + (size_t)(col0 + 16 + colL) * K;
  const _Float16* wt2 = WT + (size_t)(col0 + 32 + colL) * K;
  const _Float16* wt3 = WT + (size_t)(col0 + 48 + colL) * K;

  v8f acc0 = {}, acc1 = {}, acc2 = {}, acc3 = {};
  for (int kc = 0; kc < K; kc += 32) {
    float alo[8], ahi[8];
    ld8(ap + kc + khalf, alo);                 // K = kc+khalf+0..7
    ld8(ap + kc + khalf + 16, ahi);            // K = kc+khalf+16..23
    v16h af;
#pragma unroll
    for (int i = 0; i < 8; ++i) {
      af[i]     = (_Float16)alo[i];
      af[i + 8] = (_Float16)ahi[i];
    }
    int off = kc + kbB;
    v16h b0 = load_bfrag(wt0 + off);
    v16h b1 = load_bfrag(wt1 + off);
    v16h b2 = load_bfrag(wt2 + off);
    v16h b3 = load_bfrag(wt3 + off);
    acc0 = __builtin_amdgcn_wmma_f32_16x16x32_f16(false, af, false, b0, (short)0, acc0, false, false);
    acc1 = __builtin_amdgcn_wmma_f32_16x16x32_f16(false, af, false, b1, (short)0, acc1, false, false);
    acc2 = __builtin_amdgcn_wmma_f32_16x16x32_f16(false, af, false, b2, (short)0, acc2, false, false);
    acc3 = __builtin_amdgcn_wmma_f32_16x16x32_f16(false, af, false, b3, (short)0, acc3, false, false);
  }

  int rbase = (lane >> 4) * 8;
#pragma unroll
  for (int c = 0; c < 4; ++c) {
    v8f acc = (c == 0) ? acc0 : (c == 1) ? acc1 : (c == 2) ? acc2 : acc3;
    int col = col0 + c * 16 + colL;
    float bb = bias[col];
#pragma unroll
    for (int r = 0; r < 8; ++r) {
      int row = row0 + rbase + r;
      if (row < M) {
        size_t idx = (size_t)row * N + col;
        float x = acc[r];
        float o;
        if (MODE == 0)      o = x + bb;
        else if (MODE == 1) o = base[idx] + x + bb;
        else if (MODE == 2) o = base[idx] + x * rowscale[row] + bb;
        else { float t0 = x + bb; o = 0.5f * t0 * (1.0f + erff(t0 * 0.70710678118654752f)); }
        out[idx] = o;
      }
    }
  }
}

// ------------------------------- launcher ---------------------------------
extern "C" void kernel_launch(void* const* d_in, const int* in_sizes, int n_in,
                              void* d_out, int out_size, void* d_ws, size_t ws_size,
                              hipStream_t stream) {
  const float* h      = (const float*)d_in[0];
  const int*   e_src  = (const int*)d_in[1];
  const int*   e_dst  = (const int*)d_in[2];
  const float* pre_W  = (const float*)d_in[3];
  const float* pre_b  = (const float*)d_in[4];
  const float* post_W = (const float*)d_in[5];
  const float* post_b = (const float*)d_in[6];
  const float* q_W    = (const float*)d_in[7];
  const float* q_b    = (const float*)d_in[8];
  const float* k_W    = (const float*)d_in[9];
  const float* k_b    = (const float*)d_in[10];
  const float* v_W    = (const float*)d_in[11];
  const float* v_b    = (const float*)d_in[12];
  const float* o_W    = (const float*)d_in[13];
  const float* o_b    = (const float*)d_in[14];
  const float* f_W1   = (const float*)d_in[15];
  const float* f_b1   = (const float*)d_in[16];
  const float* f_W2   = (const float*)d_in[17];
  const float* f_b2   = (const float*)d_in[18];
  const float* lpre_g = (const float*)d_in[19];
  const float* lpre_b = (const float*)d_in[20];
  const float* latt_g = (const float*)d_in[21];
  const float* latt_b = (const float*)d_in[22];
  const float* lpost_g= (const float*)d_in[23];
  const float* lpost_b= (const float*)d_in[24];
  const float* lffn_g = (const float*)d_in[25];
  const float* lffn_b = (const float*)d_in[26];
  float* outp = (float*)d_out;

  // ---------- workspace layout (floats) ----------
  const size_t P = (size_t)NPER * HID;   // per-type activation block
  const size_t F = (size_t)NTOT * HID;   // full activation block (12.8M)
  float* ws   = (float*)d_ws;
  float* lnb  = ws;                      // F : LN output (reused each stage)
  float* hA   = lnb + F;                 // F : residual ping
  float* hB   = hA + F;                  // F : residual pong
  float* aggb = hB + F;                  // F : conv agg / attn out (reused)
  float* qb   = aggb + F;                // F : q   (f1 = qb..qb+2F)
  float* kb   = qb + F;                  // F : k
  float* vb   = kb + F;                  // F : v
  float* exb  = vb + F;                                // ET*EPER*HEADS
  float* dnb  = exb + (size_t)ET_N * EPER * HEADS;     // NTOT*HEADS
  float* csb  = dnb + (size_t)NTOT * HEADS;            // ET*NPER
  float* cdb  = csb + (size_t)ET_N * NPER;             // ET*NPER
  float* f1b  = qb;                      // FFN hidden [T,NPER,512] aliases q∪k

  // f16 transposed weights
  const size_t HH  = (size_t)HID * HID;
  const size_t H12 = (size_t)HID * HID2;
  _Float16* wtb     = (_Float16*)(cdb + (size_t)ET_N * NPER);
  _Float16* wt_pre  = wtb;
  _Float16* wt_post = wt_pre  + 2 * HH;
  _Float16* wt_q    = wt_post + 2 * HH;
  _Float16* wt_k    = wt_q    + 2 * HH;
  _Float16* wt_v    = wt_k    + 2 * HH;
  _Float16* wt_o    = wt_v    + 2 * HH;
  _Float16* wt_f1   = wt_o    + 2 * HH;
  _Float16* wt_f2   = wt_f1   + 2 * H12;

  const int MT = (NPER + 15) / 16;             // 1563 row tiles
  const int NG256 = HID / 64, NG512 = HID2 / 64;
  const int TT256 = MT * NG256, TT512 = MT * NG512;
  dim3 blkG(128), grdG((TT256 + 3) / 4), grdG2((TT512 + 3) / 4);
  dim3 blk256(256);
  dim3 grdLN(NTOT / 8);
  dim3 grdEdgeAll(((long)ET_N * EPER + 7) / 8);
  dim3 grdEdge((EPER + 7) / 8);
  auto zgrid = [](long n) { return dim3((unsigned)((n + 255) / 256)); };

  // ---------- weight pre-transpose/convert (once per launch) ----------
  wtrans_kernel<<<zgrid(2 * (long)HH),  blk256, 0, stream>>>(pre_W,  wt_pre,  HID,  HID,  2);
  wtrans_kernel<<<zgrid(2 * (long)HH),  blk256, 0, stream>>>(post_W, wt_post, HID,  HID,  2);
  wtrans_kernel<<<zgrid(2 * (long)HH),  blk256, 0, stream>>>(q_W,    wt_q,    HID,  HID,  2);
  wtrans_kernel<<<zgrid(2 * (long)HH),  blk256, 0, stream>>>(k_W,    wt_k,    HID,  HID,  2);
  wtrans_kernel<<<zgrid(2 * (long)HH),  blk256, 0, stream>>>(v_W,    wt_v,    HID,  HID,  2);
  wtrans_kernel<<<zgrid(2 * (long)HH),  blk256, 0, stream>>>(o_W,    wt_o,    HID,  HID,  2);
  wtrans_kernel<<<zgrid(2 * (long)H12), blk256, 0, stream>>>(f_W1,   wt_f1,   HID,  HID2, 2);
  wtrans_kernel<<<zgrid(2 * (long)H12), blk256, 0, stream>>>(f_W2,   wt_f2,   HID2, HID,  2);

  // ---------- degrees (shared by both convs) ----------
  long ndeg = (long)2 * ET_N * NPER;
  zero_kernel<<<zgrid(ndeg), blk256, 0, stream>>>(csb, ndeg);
  deg_kernel<<<zgrid((long)ET_N * EPER), blk256, 0, stream>>>(e_src, e_dst, csb, cdb);
  invsqrt_kernel<<<zgrid(ndeg), blk256, 0, stream>>>(csb, ndeg);

  // ---------- stage 1: h1 = h + heteroconv(LN(h), pre) ----------
  ln_kernel<<<grdLN, blk256, 0, stream>>>(h, lpre_g, lpre_b, lnb);
  zero_kernel<<<zgrid((long)F), blk256, 0, stream>>>(aggb, (long)F);
  conv_scatter_kernel<<<grdEdgeAll, blk256, 0, stream>>>(e_src, e_dst, lnb, csb, aggb);
  for (int dt = 0; dt < 2; ++dt) {
    int et = 1 - dt;                           // dst type dt fed by etype et
    gemm_kernel<2><<<grdG, blkG, 0, stream>>>(
        aggb + dt * P, wt_pre + (size_t)et * HH, pre_b + (size_t)et * HID,
        h + dt * P, cdb + (size_t)et * NPER, hA + dt * P,
        NPER, HID, HID, NG256, TT256);
  }

  // ---------- stage 2: h2 = h1 + sparse_attn(LN(h1)) ----------
  ln_kernel<<<grdLN, blk256, 0, stream>>>(hA, latt_g, latt_b, lnb);
  for (int t = 0; t < 2; ++t) {
    gemm_kernel<0><<<grdG, blkG, 0, stream>>>(lnb + t * P, wt_q + (size_t)t * HH,
        q_b + (size_t)t * HID, nullptr, nullptr, qb + t * P, NPER, HID, HID, NG256, TT256);
    gemm_kernel<0><<<grdG, blkG, 0, stream>>>(lnb + t * P, wt_k + (size_t)t * HH,
        k_b + (size_t)t * HID, nullptr, nullptr, kb + t * P, NPER, HID, HID, NG256, TT256);
    gemm_kernel<0><<<grdG, blkG, 0, stream>>>(lnb + t * P, wt_v + (size_t)t * HH,
        v_b + (size_t)t * HID, nullptr, nullptr, vb + t * P, NPER, HID, HID, NG256, TT256);
  }
  zero_kernel<<<zgrid((long)NTOT * HEADS), blk256, 0, stream>>>(dnb, (long)NTOT * HEADS);
  // etype 0: (st=0,dt=1) -> src_off 0, dst_off NPER ; etype 1: reversed
  attn_score_kernel<<<grdEdge, blk256, 0, stream>>>(e_src, e_dst, qb, kb,
      exb, dnb, 0, NPER);
  attn_score_kernel<<<grdEdge, blk256, 0, stream>>>(e_src + EPER, e_dst + EPER, qb, kb,
      exb + (size_t)EPER * HEADS, dnb, NPER, 0);
  zero_kernel<<<zgrid((long)F), blk256, 0, stream>>>(aggb, (long)F);
  attn_agg_kernel<<<grdEdge, blk256, 0, stream>>>(e_src, e_dst, exb, dnb, vb, aggb, 0, NPER);
  attn_agg_kernel<<<grdEdge, blk256, 0, stream>>>(e_src + EPER, e_dst + EPER,
      exb + (size_t)EPER * HEADS, dnb, vb, aggb, NPER, 0);
  for (int t = 0; t < 2; ++t) {
    gemm_kernel<1><<<grdG, blkG, 0, stream>>>(aggb + t * P, wt_o + (size_t)t * HH,
        o_b + (size_t)t * HID, hA + t * P, nullptr, hB + t * P, NPER, HID, HID, NG256, TT256);
  }

  // ---------- stage 3: h3 = h2 + heteroconv(LN(h2), post) ----------
  ln_kernel<<<grdLN, blk256, 0, stream>>>(hB, lpost_g, lpost_b, lnb);
  zero_kernel<<<zgrid((long)F), blk256, 0, stream>>>(aggb, (long)F);
  conv_scatter_kernel<<<grdEdgeAll, blk256, 0, stream>>>(e_src, e_dst, lnb, csb, aggb);
  for (int dt = 0; dt < 2; ++dt) {
    int et = 1 - dt;
    gemm_kernel<2><<<grdG, blkG, 0, stream>>>(
        aggb + dt * P, wt_post + (size_t)et * HH, post_b + (size_t)et * HID,
        hB + dt * P, cdb + (size_t)et * NPER, hA + dt * P,
        NPER, HID, HID, NG256, TT256);
  }

  // ---------- stage 4: out = h3 + FFN(LN(h3)) ----------
  ln_kernel<<<grdLN, blk256, 0, stream>>>(hA, lffn_g, lffn_b, lnb);
  for (int t = 0; t < 2; ++t) {
    gemm_kernel<3><<<grdG2, blkG, 0, stream>>>(lnb + t * P,
        wt_f1 + (size_t)t * H12, f_b1 + (size_t)t * HID2, nullptr, nullptr,
        f1b + (size_t)t * NPER * HID2, NPER, HID, HID2, NG512, TT512);
  }
  for (int t = 0; t < 2; ++t) {
    gemm_kernel<1><<<grdG, blkG, 0, stream>>>(f1b + (size_t)t * NPER * HID2,
        wt_f2 + (size_t)t * H12, f_b2 + (size_t)t * HID, hA + t * P, nullptr,
        outp + t * P, NPER, HID2, HID, NG256, TT256);
  }
}
